// RNN_7894149890770
// MI455X (gfx1250) — matrix operand order repaired and checked by reading, not compiled, not run on hardware
//
#include <hip/hip_runtime.h>

typedef __attribute__((ext_vector_type(16))) _Float16 v16h;
typedef __attribute__((ext_vector_type(8)))  _Float16 v8h;
typedef __attribute__((ext_vector_type(8)))  float    v8f;

#define SEQ 131072
#define ISZ 100
#define HSZ 40
#define OSZ 100

// ---------------------------------------------------------------------------
// Stage a zero-padded f16 tile into LDS. Branch-free: clamped address + select.
// dst_ld must be a power of two (compile-time constant at call sites).
// ---------------------------------------------------------------------------
__device__ __forceinline__ void stage_tile_f16(
    const float* __restrict__ src, int src_ld, int rows_valid, int cols_valid,
    _Float16* dst, int dst_ld, int total, int tid, int nthreads) {
  for (int idx = tid; idx < total; idx += nthreads) {
    const int r = idx / dst_ld;          // folds to shift (dst_ld is 2^k)
    const int c = idx - r * dst_ld;
    const bool ok = (r < rows_valid) && (c < cols_valid);
    const long off = ok ? ((long)r * src_ld + c) : 0;
    const float v = src[off];            // unconditional, clamped address
    dst[idx] = (_Float16)(ok ? v : 0.0f);
  }
}

// Build a v16h A-fragment from two contiguous 8-f16 runs in LDS
// (elements 0..7 -> K=koff+0..7, elements 8..15 -> K=koff+8..15; contiguous k=16h+s, matching B)
__device__ __forceinline__ v16h frag_from_runs(const _Float16* p0,
                                               const _Float16* p1) {
  const v8h lo = *(const v8h*)p0;        // ds_load_b128
  const v8h hi = *(const v8h*)p1;        // ds_load_b128
  v16h r;
#pragma unroll
  for (int e = 0; e < 8; ++e) { r[e] = lo[e]; r[8 + e] = hi[e]; }
  return r;
}

// ---------------------------------------------------------------------------
// Phase 1: xsb[t][i] = (s @ Wx_w.T)[t][i] + Wx_b[i] + Wh_b[i]
//   128 threads (4 waves) per 64 timesteps. K=100 -> 4 chunks of 32 (pad 128).
//   N=40 -> 3 tiles of 16 (pad 48). 12 WMMAs per wave, all operands from LDS.
// ---------------------------------------------------------------------------
__global__ __launch_bounds__(128) void rnn_xproj_kernel(
    const float* __restrict__ s, const float* __restrict__ Wx_w,
    const float* __restrict__ Wx_b, const float* __restrict__ Wh_b,
    float* __restrict__ xsb) {
  __shared__ __align__(32) _Float16 lds_a[64][128];   // 16 KB
  __shared__ __align__(32) _Float16 lds_b[48][128];   // 12 KB
  const int  tid  = threadIdx.x;
  const int  lane = tid & 31;
  const int  wave = tid >> 5;
  const long t0   = (long)blockIdx.x * 64;

  stage_tile_f16(s + t0 * ISZ, ISZ, 64, ISZ, &lds_a[0][0], 128, 64 * 128, tid, 128);
  stage_tile_f16(Wx_w,         ISZ, HSZ, ISZ, &lds_b[0][0], 128, 48 * 128, tid, 128);
  __syncthreads();

  const int m     = lane & 15;
  const int khalf = (lane < 16) ? 0 : 8;
  const int nn    = lane & 15;
  const int koff  = (lane < 16) ? 0 : 16;
  const _Float16* arow = &lds_a[wave * 16 + m][0];

  v8f c0 = {}, c1 = {}, c2 = {};
#pragma unroll
  for (int kc = 0; kc < 4; ++kc) {
    const int kb = kc * 32;
    v16h a  = frag_from_runs(arow + kb + koff, arow + kb + koff + 8);
    v16h b0 = *(const v16h*)&lds_b[nn +  0][kb + koff];
    v16h b1 = *(const v16h*)&lds_b[nn + 16][kb + koff];
    v16h b2 = *(const v16h*)&lds_b[nn + 32][kb + koff];
    c0 = __builtin_amdgcn_wmma_f32_16x16x32_f16(false, a, false, b0, (short)0, c0, false, false);
    c1 = __builtin_amdgcn_wmma_f32_16x16x32_f16(false, a, false, b1, (short)0, c1, false, false);
    c2 = __builtin_amdgcn_wmma_f32_16x16x32_f16(false, a, false, b2, (short)0, c2, false, false);
  }

  // Epilogue: bias + scatter (C layout: VGPR r -> M = r + (lane<16?0:8), N = nn)
  const long row0 = t0 + wave * 16;
  const int  mb   = (lane < 16) ? 0 : 8;
#pragma unroll
  for (int tn = 0; tn < 3; ++tn) {
    const int n  = tn * 16 + nn;
    const int nc = (n < HSZ) ? n : 0;                 // clamp, no exec branch
    const float bias = Wx_b[nc] + Wh_b[nc];
    const v8f c = (tn == 0) ? c0 : (tn == 1) ? c1 : c2;
    if (n < HSZ) {
#pragma unroll
      for (int r = 0; r < 8; ++r)
        xsb[(row0 + mb + r) * (long)HSZ + n] = c[r] + bias;
    }
  }
}

// ---------------------------------------------------------------------------
// Phase 2: serial 131072-step scan (latency-bound; tanh blocks parallel scan).
//   One block, 40 active lanes. Wh row register-resident; h broadcast via LDS
//   read as float4 (ds_load_b128); 4-way accumulators cut the FMA chain to
//   ~10 deep; upcoming xs rows prefetched (global_prefetch_b8).
// ---------------------------------------------------------------------------
__global__ __launch_bounds__(64) void rnn_scan_kernel(
    const float* __restrict__ xsb, const float* __restrict__ h0,
    const float* __restrict__ Wh_w, float* __restrict__ Hout,
    float* __restrict__ hfinal, int seq) {
  __shared__ __align__(16) float hs[HSZ];
  const int i = threadIdx.x;

  float wrow[HSZ];
  if (i < HSZ) {
#pragma unroll
    for (int j = 0; j < HSZ; ++j) wrow[j] = Wh_w[i * HSZ + j];
    hs[i] = h0[i];
  }
  __syncthreads();

  for (int t = 0; t < seq; ++t) {
    float hn = 0.0f;
    if (i < HSZ) {
      const float xv = xsb[(long)t * HSZ + i];        // issue early
      float a0 = 0.f, a1 = 0.f, a2 = 0.f, a3 = 0.f;
#pragma unroll
      for (int j = 0; j < HSZ; j += 4) {
        const float4 hv = *(const float4*)&hs[j];     // ds_load_b128 broadcast
        a0 += wrow[j + 0] * hv.x;
        a1 += wrow[j + 1] * hv.y;
        a2 += wrow[j + 2] * hv.z;
        a3 += wrow[j + 3] * hv.w;
      }
      hn = tanhf(xv + ((a0 + a1) + (a2 + a3)));
    }
    __syncthreads();
    if (i < HSZ) {
      hs[i] = hn;
      Hout[(long)t * HSZ + i] = hn;                   // no dependence, hidden
    }
    if (i == 0 && (t + 8) < seq)
      __builtin_prefetch(&xsb[(long)(t + 8) * HSZ], 0, 0);
    __syncthreads();
  }

  if (i < HSZ) hfinal[i] = hs[i];
}

// ---------------------------------------------------------------------------
// Phase 3: ys = softmax(H @ Wy_w.T + Wy_b) per row.
//   128 threads per 64 timesteps; each wave owns 16 rows, loops 7 N-tiles
//   (N=100 pad 112) x 2 K-chunks (K=40 pad 64) = 14 WMMAs per wave.
//   Logits staged in LDS; 64 threads then softmax one row each.
// ---------------------------------------------------------------------------
__global__ __launch_bounds__(128) void rnn_out_kernel(
    const float* __restrict__ H, const float* __restrict__ Wy_w,
    const float* __restrict__ Wy_b, float* __restrict__ ys) {
  __shared__ __align__(32) _Float16 lds_a[64][64];    //  8 KB
  __shared__ __align__(32) _Float16 lds_b[112][64];   // 14 KB
  __shared__ float logits[64][112];                   // 28 KB
  const int  tid  = threadIdx.x;
  const int  lane = tid & 31;
  const int  wave = tid >> 5;
  const long t0   = (long)blockIdx.x * 64;

  stage_tile_f16(H + t0 * HSZ, HSZ, 64, HSZ, &lds_a[0][0], 64,  64 * 64, tid, 128);
  stage_tile_f16(Wy_w,         HSZ, OSZ, HSZ, &lds_b[0][0], 64, 112 * 64, tid, 128);
  __syncthreads();

  const int m     = lane & 15;
  const int khalf = (lane < 16) ? 0 : 8;
  const int nn    = lane & 15;
  const int koff  = (lane < 16) ? 0 : 16;
  const int mb    = (lane < 16) ? 0 : 8;
  const _Float16* arow = &lds_a[wave * 16 + m][0];

#pragma unroll
  for (int tn = 0; tn < 7; ++tn) {
    v8f c = {};
#pragma unroll
    for (int kc = 0; kc < 2; ++kc) {
      const int kb = kc * 32;
      v16h a = frag_from_runs(arow + kb + koff, arow + kb + koff + 8);
      v16h b = *(const v16h*)&lds_b[tn * 16 + nn][kb + koff];
      c = __builtin_amdgcn_wmma_f32_16x16x32_f16(false, a, false, b, (short)0, c, false, false);
    }
    const int n  = tn * 16 + nn;
    const int nc = (n < OSZ) ? n : 0;
    const float bias = (n < OSZ) ? Wy_b[nc] : 0.0f;
#pragma unroll
    for (int r = 0; r < 8; ++r)
      logits[wave * 16 + mb + r][n] = c[r] + bias;
  }
  __syncthreads();

  if (tid < 64) {
    float* row = logits[tid];
    float mx = row[0];
#pragma unroll 4
    for (int c = 1; c < OSZ; ++c) mx = fmaxf(mx, row[c]);
    float sum = 0.0f;
#pragma unroll 4
    for (int c = 0; c < OSZ; ++c) {
      const float e = __expf(row[c] - mx);
      row[c] = e;
      sum += e;
    }
    const float inv = 1.0f / sum;
    float* orow = ys + (t0 + tid) * OSZ;
#pragma unroll 4
    for (int c = 0; c < OSZ; ++c) orow[c] = row[c] * inv;
  }
}

// ---------------------------------------------------------------------------
extern "C" void kernel_launch(void* const* d_in, const int* in_sizes, int n_in,
                              void* d_out, int out_size, void* d_ws, size_t ws_size,
                              hipStream_t stream) {
  const float* s    = (const float*)d_in[0];
  const float* h0   = (const float*)d_in[1];
  const float* Wx_w = (const float*)d_in[2];
  const float* Wx_b = (const float*)d_in[3];
  const float* Wh_w = (const float*)d_in[4];
  const float* Wh_b = (const float*)d_in[5];
  const float* Wy_w = (const float*)d_in[6];
  const float* Wy_b = (const float*)d_in[7];

  float* out    = (float*)d_out;
  float* hfinal = out;         // tuple element 0: h_final (40)
  float* ys     = out + HSZ;   // tuple element 1: ys [SEQ,100]

  float* xsb  = (float*)d_ws;                 // [SEQ, 40]
  float* Hbuf = xsb + (size_t)SEQ * HSZ;      // [SEQ, 40]

  rnn_xproj_kernel<<<SEQ / 64, 128, 0, stream>>>(s, Wx_w, Wx_b, Wh_b, xsb);
  rnn_scan_kernel<<<1, 64, 0, stream>>>(xsb, h0, Wh_w, Hbuf, hfinal, SEQ);
  rnn_out_kernel<<<SEQ / 64, 128, 0, stream>>>(Hbuf, Wy_w, Wy_b, ys);
}
